// TransDecSamplingModel_64390149701919
// MI455X (gfx1250) — compile-verified, matrix operand order checked by reading, not run on hardware
//
#include <hip/hip_runtime.h>
#include <hip/hip_bf16.h>

typedef __attribute__((ext_vector_type(16))) _Float16 v16h;
typedef __attribute__((ext_vector_type(8)))  float    v8f;

#define D        512
#define VOCAB1   8193   // VOCAB+1
#define VPAD     8704   // VOCAB1 padded to multiple of 512 (17*512)
#define UTT      16
#define BATCH    32
#define NTYPES   8
#define MPT      32
#define QKVS     1536
#define KCHUNK   128
#define ASTRIDE  (KCHUNK + 8)   // LDS row stride in halves: 272B/row -> conflict-free b128 reads

// ---------------------------------------------------------------------------
// fp32 -> f16 weight conversion (once per launch; weights then live in L2)
// ---------------------------------------------------------------------------
__global__ void cvt_f16_kernel(const float* __restrict__ s, _Float16* __restrict__ d, size_t n) {
  size_t i = (size_t)blockIdx.x * blockDim.x + threadIdx.x;
  if (i < n) d[i] = (_Float16)s[i];
}

__global__ void cvt_f16_pad_rows_kernel(const float* __restrict__ s, _Float16* __restrict__ d,
                                        int rows, int padRows, int cols) {
  size_t i = (size_t)blockIdx.x * blockDim.x + threadIdx.x;
  size_t tot = (size_t)padRows * cols;
  if (i < tot) {
    int r = (int)(i / cols);
    d[i] = (r < rows) ? (_Float16)s[i] : (_Float16)0.0f;
  }
}

__global__ void init_buf_kernel(int* __restrict__ buf, int n, int Bsz, int start) {
  int i = blockIdx.x * blockDim.x + threadIdx.x;
  if (i < n) buf[i] = (i < Bsz) ? start : 0;
}

// src[b,:] = sum_t emb_table[meanings[b,t] + t*MPT, :]
__global__ void build_src_kernel(const int* __restrict__ meanings, const float* __restrict__ emb,
                                 float* __restrict__ src) {
  int b = blockIdx.x;
  for (int dd = threadIdx.x; dd < D; dd += blockDim.x) {
    float s = 0.0f;
    #pragma unroll
    for (int t = 0; t < NTYPES; ++t) {
      int idx = meanings[b * NTYPES + t] + t * MPT;
      s += emb[(size_t)idx * D + dd];
    }
    src[(size_t)b * D + dd] = s;
  }
}

// x[t,b,:] = v2e_w[buf[t,b],:] + sinusoidal_pe(t,:)
__global__ void embed_kernel(const int* __restrict__ buf, const float* __restrict__ v2e,
                             float* __restrict__ x) {
  int t = blockIdx.x / BATCH, b = blockIdx.x % BATCH;
  int tok = buf[t * BATCH + b];
  const float* e = v2e + (size_t)tok * D;
  float* xp = x + (size_t)(t * BATCH + b) * D;
  for (int dd = threadIdx.x; dd < D; dd += blockDim.x) {
    int i2 = dd & ~1;
    float div = __expf(-9.21034037f * (float)i2 * (1.0f / (float)D)); // ln(10000)
    float ang = (float)t * div;
    float pe = (dd & 1) ? __cosf(ang) : __sinf(ang);
    xp[dd] = e[dd] + pe;
  }
}

// ---------------------------------------------------------------------------
// WMMA GEMM:  Y[R,N] = act( A[R,K](f32) @ W[N,K](f16)^T + bias )
// Block = 8 waves = one 16-row M strip x 512 columns.  Each wave: 16x64 tile
// (4 accumulators -> 4 WMMAs per A fragment).  A strip staged via LDS in f16
// once per 128-deep K chunk (8x less global A traffic + conversion).
// All 4 B fragments are batch-loaded before the 4 WMMAs so the hardware can
// use partial s_wait_loadcnt values (overlap loads with matrix ops).
// R % 16 == 0, N % 512 == 0, K % 128 == 0.
// ---------------------------------------------------------------------------
__global__ void gemm_f16w_kernel(const float* __restrict__ A,
                                 const _Float16* __restrict__ Bw,
                                 const float* __restrict__ bias, int biasN,
                                 float* __restrict__ Y,
                                 int R, int K, int N, int relu) {
  const int lane  = threadIdx.x & 31;
  const int wave  = threadIdx.x >> 5;
  const int mt    = blockIdx.x;                     // 16-row strip
  const int n0    = (blockIdx.y * 8 + wave) * 64;   // wave's 64-col group
  const int rl    = lane & 15;                      // M (A) / N (B) within 16
  const int kbase = (lane >> 4) << 3;               // 0 or 8

  __shared__ _Float16 As[16 * ASTRIDE];

  const float* Ab = A + (size_t)mt * 16 * K;
  v8f acc[4] = {{}, {}, {}, {}};

  for (int k0 = 0; k0 < K; k0 += KCHUNK) {
    // ---- cooperative stage: A[16, k0:k0+128] -> LDS (f16) ----
    #pragma unroll
    for (int i = 0; i < (16 * KCHUNK) / 256; ++i) {
      int e = threadIdx.x + i * 256;
      int r = e >> 7;               // /KCHUNK
      int c = e & (KCHUNK - 1);
      As[r * ASTRIDE + c] = (_Float16)Ab[(size_t)r * K + k0 + c];
    }
    __syncthreads();

    #pragma unroll
    for (int ks = 0; ks < KCHUNK; ks += 32) {
      // A fragment from LDS (16-bit A layout: lanes 16-31 hold K+8 / K+24)
      v16h af;
      const _Float16* ar = &As[rl * ASTRIDE + ks + kbase];
      #pragma unroll
      for (int j = 0; j < 4; ++j) {
        af[2 * j]         = ar[2 * j];
        af[2 * j + 1]     = ar[2 * j + 1];
        af[8 + 2 * j]     = ar[16 + 2 * j];
        af[8 + 2 * j + 1] = ar[16 + 2 * j + 1];
      }
      const int kg = k0 + ks + kbase;
      // batch-load all 4 B fragments (8 b128 loads in flight), then 4 WMMAs
      v16h bfr[4];
      #pragma unroll
      for (int t = 0; t < 4; ++t) {
        const _Float16* br = Bw + (size_t)(n0 + t * 16 + rl) * K + kg;
        if (ks == 0 && k0 + KCHUNK < K)             // warm next B chunk
          __builtin_prefetch((const void*)(br + KCHUNK), 0, 3);
        #pragma unroll
        for (int j = 0; j < 4; ++j) {
          bfr[t][2 * j]         = br[2 * j];
          bfr[t][2 * j + 1]     = br[2 * j + 1];
          bfr[t][8 + 2 * j]     = br[16 + 2 * j];
          bfr[t][8 + 2 * j + 1] = br[16 + 2 * j + 1];
        }
      }
      #pragma unroll
      for (int t = 0; t < 4; ++t)
        acc[t] = __builtin_amdgcn_wmma_f32_16x16x32_f16(false, af, false, bfr[t],
                                                        (short)0, acc[t], false, false);
    }
    __syncthreads();
  }

  // ---- store: C/D layout -> lane rl = column, VGPR v = row (lanes 16-31: +8)
  #pragma unroll
  for (int t = 0; t < 4; ++t) {
    const int n = n0 + t * 16 + rl;
    float bv = bias ? ((n < biasN) ? bias[n] : 0.0f) : 0.0f;
    #pragma unroll
    for (int v = 0; v < 8; ++v) {
      int row = mt * 16 + v + kbase;
      float y = acc[t][v] + bv;
      if (relu) y = fmaxf(y, 0.0f);
      Y[(size_t)row * N + n] = y;
    }
  }
}

// ---------------------------------------------------------------------------
// causal single-head self-attention over L=16, one block per (t,b)
// qkv rows: [(t*B+b), 1536] with q@0, k@512, v@1024
// ---------------------------------------------------------------------------
__global__ void self_attn_kernel(const float* __restrict__ qkv, float* __restrict__ o) {
  const int t = blockIdx.x / BATCH, b = blockIdx.x % BATCH;
  __shared__ float red[256];
  __shared__ float sc[UTT];
  __shared__ float pr[UTT];
  const float scale = rsqrtf((float)D);
  const float* qp = qkv + (size_t)(t * BATCH + b) * QKVS;

  for (int kk = 0; kk < UTT; ++kk) {
    const float* kp = qkv + (size_t)(kk * BATCH + b) * QKVS + D;
    float p = 0.0f;
    for (int dd = threadIdx.x; dd < D; dd += blockDim.x) p += qp[dd] * kp[dd];
    red[threadIdx.x] = p; __syncthreads();
    for (int s = 128; s > 0; s >>= 1) {
      if (threadIdx.x < s) red[threadIdx.x] += red[threadIdx.x + s];
      __syncthreads();
    }
    if (threadIdx.x == 0) sc[kk] = (kk <= t) ? red[0] * scale : -3.0e38f;
    __syncthreads();
  }
  if (threadIdx.x == 0) {
    float mx = sc[0];
    for (int kk = 1; kk < UTT; ++kk) mx = fmaxf(mx, sc[kk]);
    float s = 0.0f;
    for (int kk = 0; kk < UTT; ++kk) { float e = __expf(sc[kk] - mx); pr[kk] = e; s += e; }
    float inv = 1.0f / s;
    for (int kk = 0; kk < UTT; ++kk) pr[kk] *= inv;
  }
  __syncthreads();
  for (int dd = threadIdx.x; dd < D; dd += blockDim.x) {
    float acc = 0.0f;
    for (int kk = 0; kk <= t; ++kk)
      acc += pr[kk] * qkv[(size_t)(kk * BATCH + b) * QKVS + 2 * D + dd];
    o[(size_t)(t * BATCH + b) * D + dd] = acc;
  }
}

// ---------------------------------------------------------------------------
// x[r,:] = LN(x[r,:] + res[rr,:]) * g + b   (rr = r%B when resPerB, else r)
// ---------------------------------------------------------------------------
__global__ void add_ln_kernel(float* __restrict__ x, const float* __restrict__ res, int resPerB,
                              const float* __restrict__ g, const float* __restrict__ bb) {
  const int r = blockIdx.x;
  const int rr = resPerB ? (r % BATCH) : r;
  const float* rp = res + (size_t)rr * D;
  float* xp = x + (size_t)r * D;
  __shared__ float red[256];
  float vals[2]; float s1 = 0.0f, s2 = 0.0f;
  #pragma unroll
  for (int i = 0; i < 2; ++i) {
    int dd = threadIdx.x + i * 256;
    float v = xp[dd] + rp[dd];
    vals[i] = v; s1 += v; s2 += v * v;
  }
  red[threadIdx.x] = s1; __syncthreads();
  for (int s = 128; s > 0; s >>= 1) { if (threadIdx.x < s) red[threadIdx.x] += red[threadIdx.x + s]; __syncthreads(); }
  float mean = red[0] * (1.0f / D); __syncthreads();
  red[threadIdx.x] = s2; __syncthreads();
  for (int s = 128; s > 0; s >>= 1) { if (threadIdx.x < s) red[threadIdx.x] += red[threadIdx.x + s]; __syncthreads(); }
  float var = red[0] * (1.0f / D) - mean * mean;
  float rstd = rsqrtf(var + 1e-5f);
  #pragma unroll
  for (int i = 0; i < 2; ++i) {
    int dd = threadIdx.x + i * 256;
    xp[dd] = (vals[i] - mean) * rstd * g[dd] + bb[dd];
  }
}

// ---------------------------------------------------------------------------
// per-b argmax over logits, feed token back into buf, emit outputs
// ---------------------------------------------------------------------------
__global__ void argmax_kernel(const float* __restrict__ lpad,
                              int* __restrict__ buf, int step,
                              float* __restrict__ outTok, float* __restrict__ outLogits) {
  const int b = blockIdx.x;
  const float* lp = lpad + (size_t)b * VPAD;
  __shared__ float bv[256]; __shared__ int bi[256];
  float mv = -3.4e38f; int mi = 0;
  for (int n = threadIdx.x; n < VOCAB1; n += blockDim.x) {
    float v = lp[n];
    if (v > mv) { mv = v; mi = n; }
  }
  bv[threadIdx.x] = mv; bi[threadIdx.x] = mi; __syncthreads();
  for (int s = 128; s > 0; s >>= 1) {
    if (threadIdx.x < s) {
      float ov = bv[threadIdx.x + s]; int oi = bi[threadIdx.x + s];
      if (ov > bv[threadIdx.x] || (ov == bv[threadIdx.x] && oi < bi[threadIdx.x])) {
        bv[threadIdx.x] = ov; bi[threadIdx.x] = oi;
      }
    }
    __syncthreads();
  }
  if (threadIdx.x == 0) {
    int tok = bi[0];
    buf[(step + 1) * BATCH + b] = tok;
    outTok[step * BATCH + b] = (float)tok;
  }
  for (int n = threadIdx.x; n < VOCAB1; n += blockDim.x)
    outLogits[((size_t)step * BATCH + b) * VOCAB1 + n] = lp[n];
}

// ---------------------------------------------------------------------------
extern "C" void kernel_launch(void* const* d_in, const int* in_sizes, int n_in,
                              void* d_out, int out_size, void* d_ws, size_t ws_size,
                              hipStream_t stream) {
  (void)in_sizes; (void)n_in; (void)out_size; (void)ws_size;
  const int*   meanings  = (const int*)  d_in[0];
  const float* emb_table = (const float*)d_in[1];
  const float* v2e_w     = (const float*)d_in[2];
  const float* e2v_w     = (const float*)d_in[3];
  const float* e2v_b     = (const float*)d_in[4];
  const float* sa_qkv_w  = (const float*)d_in[5];
  const float* sa_qkv_b  = (const float*)d_in[6];
  const float* sa_out_w  = (const float*)d_in[7];
  const float* sa_out_b  = (const float*)d_in[8];
  const float* ca_qkv_w  = (const float*)d_in[9];
  const float* ca_qkv_b  = (const float*)d_in[10];
  const float* ca_out_w  = (const float*)d_in[11];
  const float* ca_out_b  = (const float*)d_in[12];
  const float* ffn_w1    = (const float*)d_in[13];
  const float* ffn_b1    = (const float*)d_in[14];
  const float* ffn_w2    = (const float*)d_in[15];
  const float* ffn_b2    = (const float*)d_in[16];
  const float* ln1_g = (const float*)d_in[17]; const float* ln1_b = (const float*)d_in[18];
  const float* ln2_g = (const float*)d_in[19]; const float* ln2_b = (const float*)d_in[20];
  const float* ln3_g = (const float*)d_in[21]; const float* ln3_b = (const float*)d_in[22];
  float* out = (float*)d_out;

  // bump allocator over d_ws
  char* wsb = (char*)d_ws; size_t off = 0;
  auto alloc = [&](size_t bytes) -> void* {
    void* p = wsb + off; off += (bytes + 255) & ~(size_t)255; return p;
  };
  const int RL = UTT * BATCH; // 512 activation rows
  int*   buf      = (int*)  alloc((UTT + 1) * BATCH * sizeof(int));
  float* src      = (float*)alloc((size_t)BATCH * D * sizeof(float));
  float* x        = (float*)alloc((size_t)RL * D * sizeof(float));
  float* qkv      = (float*)alloc((size_t)RL * QKVS * sizeof(float));
  float* sa_o     = (float*)alloc((size_t)RL * D * sizeof(float));
  float* tmp      = (float*)alloc((size_t)RL * D * sizeof(float));
  float* h1       = (float*)alloc((size_t)RL * 4 * D * sizeof(float));
  float* vsrc     = (float*)alloc((size_t)BATCH * D * sizeof(float));
  float* ca_o     = (float*)alloc((size_t)BATCH * D * sizeof(float));
  float* lpad     = (float*)alloc((size_t)BATCH * VPAD * sizeof(float));
  _Float16* e2v_h    = (_Float16*)alloc((size_t)VPAD * D * sizeof(_Float16));
  _Float16* sa_qkv_h = (_Float16*)alloc((size_t)2 * QKVS * D * sizeof(_Float16));
  _Float16* sa_out_h = (_Float16*)alloc((size_t)2 * D * D * sizeof(_Float16));
  _Float16* ca_v_h   = (_Float16*)alloc((size_t)2 * D * D * sizeof(_Float16));
  _Float16* ca_out_h = (_Float16*)alloc((size_t)2 * D * D * sizeof(_Float16));
  _Float16* ffn_w1_h = (_Float16*)alloc((size_t)2 * 4 * D * D * sizeof(_Float16));
  _Float16* ffn_w2_h = (_Float16*)alloc((size_t)2 * D * 4 * D * sizeof(_Float16));

  auto cvt = [&](const float* s, _Float16* d, size_t n) {
    cvt_f16_kernel<<<(unsigned)((n + 255) / 256), 256, 0, stream>>>(s, d, n);
  };
  auto gemm = [&](const float* A, const _Float16* Bw, const float* bias, int biasN,
                  float* Y, int R, int K, int N, int relu) {
    dim3 grid(R / 16, N / 512);
    gemm_f16w_kernel<<<grid, 256, 0, stream>>>(A, Bw, bias, biasN, Y, R, K, N, relu);
  };

  // ---- one-time setup (per launch; deterministic) ----
  init_buf_kernel<<<((UTT + 1) * BATCH + 255) / 256, 256, 0, stream>>>(
      buf, (UTT + 1) * BATCH, BATCH, VOCAB1 - 1);
  build_src_kernel<<<BATCH, 256, 0, stream>>>(meanings, emb_table, src);
  cvt(sa_qkv_w, sa_qkv_h, (size_t)2 * QKVS * D);
  cvt(sa_out_w, sa_out_h, (size_t)2 * D * D);
  cvt(ca_out_w, ca_out_h, (size_t)2 * D * D);
  cvt(ffn_w1,   ffn_w1_h, (size_t)2 * 4 * D * D);
  cvt(ffn_w2,   ffn_w2_h, (size_t)2 * D * 4 * D);
  for (int l = 0; l < 2; ++l)  // only the V slice of cross-attn QKV is ever used
    cvt(ca_qkv_w + ((size_t)l * QKVS + 2 * D) * D, ca_v_h + (size_t)l * D * D, (size_t)D * D);
  {
    size_t tot = (size_t)VPAD * D;
    cvt_f16_pad_rows_kernel<<<(unsigned)((tot + 255) / 256), 256, 0, stream>>>(
        e2v_w, e2v_h, VOCAB1, VPAD, D);
  }

  // ---- 16 strictly sequential decode steps ----
  for (int i = 0; i < UTT; ++i) {
    embed_kernel<<<RL, 256, 0, stream>>>(buf, v2e_w, x);
    for (int l = 0; l < 2; ++l) {
      gemm(x, sa_qkv_h + (size_t)l * QKVS * D, sa_qkv_b + l * QKVS, QKVS,
           qkv, RL, D, QKVS, 0);
      self_attn_kernel<<<RL, 256, 0, stream>>>(qkv, sa_o);
      gemm(sa_o, sa_out_h + (size_t)l * D * D, sa_out_b + l * D, D,
           tmp, RL, D, D, 0);
      add_ln_kernel<<<RL, 256, 0, stream>>>(x, tmp, 0, ln1_g + l * D, ln1_b + l * D);
      // cross-attn with 1-token memory == broadcast out_proj(v_proj(src))
      gemm(src, ca_v_h + (size_t)l * D * D, ca_qkv_b + l * QKVS + 2 * D, D,
           vsrc, BATCH, D, D, 0);
      gemm(vsrc, ca_out_h + (size_t)l * D * D, ca_out_b + l * D, D,
           ca_o, BATCH, D, D, 0);
      add_ln_kernel<<<RL, 256, 0, stream>>>(x, ca_o, 1, ln2_g + l * D, ln2_b + l * D);
      gemm(x, ffn_w1_h + (size_t)l * 4 * D * D, ffn_b1 + l * 4 * D, 4 * D,
           h1, RL, D, 4 * D, 1);
      gemm(h1, ffn_w2_h + (size_t)l * D * 4 * D, ffn_b2 + l * D, D,
           tmp, RL, 4 * D, D, 0);
      add_ln_kernel<<<RL, 256, 0, stream>>>(x, tmp, 0, ln3_g + l * D, ln3_b + l * D);
    }
    // logits for position i only: rows [i*B, i*B+32)
    gemm(x + (size_t)i * BATCH * D, e2v_h, e2v_b, VOCAB1, lpad, BATCH, D, VPAD, 0);
    argmax_kernel<<<BATCH, 256, 0, stream>>>(lpad, buf, i, out, out + UTT * BATCH);
  }
}
